// LocalAtomAttention_56023553409230
// MI455X (gfx1250) — matrix operand order repaired and checked by reading, not compiled, not run on hardware
//
#include <hip/hip_runtime.h>
#include <hip/hip_bf16.h>

typedef __attribute__((ext_vector_type(16))) _Float16 v16h;
typedef __attribute__((ext_vector_type(8)))  _Float16 v8h;
typedef __attribute__((ext_vector_type(8)))  float    v8f;

#define CDIM 128
#define HD   32
#define TILE 16
#define WAVES 4

static __device__ inline v8f wmma_f16(v16h a, v16h b, v8f c) {
  // D = A(16x32 f16) * B(32x16 f16) + C(16x16 f32)
  return __builtin_amdgcn_wmma_f32_16x16x32_f16(false, a, false, b, (short)0, c,
                                                false, false);
}

// Convert + transpose the four 128x128 fp32 weight matrices into f16:
// Wt[mat][n][k] = (f16) W[mat][k][n]   (so B-fragments load contiguously)
__global__ void prep_weights_kernel(const float* __restrict__ Wq,
                                    const float* __restrict__ Wk,
                                    const float* __restrict__ Wv,
                                    const float* __restrict__ Wo,
                                    _Float16* __restrict__ Wt) {
  int idx = blockIdx.x * blockDim.x + threadIdx.x;
  if (idx >= 4 * CDIM * CDIM) return;
  int mat = idx >> 14;
  int rem = idx & (CDIM * CDIM - 1);
  int n = rem >> 7;
  int k = rem & (CDIM - 1);
  const float* W = (mat == 0) ? Wq : (mat == 1) ? Wk : (mat == 2) ? Wv : Wo;
  Wt[idx] = (_Float16)W[k * CDIM + n];
}

__global__ __launch_bounds__(WAVES * 32) void fused_local_attn(
    const float* __restrict__ x, const int* __restrict__ mask,
    const _Float16* __restrict__ Wt,
    const float* __restrict__ bq, const float* __restrict__ bk,
    const float* __restrict__ bv, const float* __restrict__ bo,
    float* __restrict__ out, int ntiles) {
  // per-wave staging: q(2048) k(2048) v(2048) attended(2048) halves = 16 KB
  __shared__ _Float16 lds[WAVES][8192];

  const int tid  = threadIdx.x;
  const int wave = tid >> 5;
  const int lane = tid & 31;
  const int half = lane >> 4;   // lane group: 0 -> lanes 0-15, 1 -> 16-31
  const int hn   = lane & 15;

  const int tile = blockIdx.x * WAVES + wave;
  if (tile >= ntiles) return;
  const int base = tile * TILE;  // first flat token (b,l,a) of tile

  _Float16* qb = &lds[wave][0];
  _Float16* kb = qb + 2048;
  _Float16* vb = qb + 4096;
  _Float16* ab = qb + 6144;

  // ---- hoisted per-lane biases: bias[mat][ntile] at channel ntile*16+hn ----
  float bias[4][8];
  const float* bptr[4] = {bq, bk, bv, bo};
  #pragma unroll
  for (int m = 0; m < 4; ++m)
    #pragma unroll
    for (int nt = 0; nt < 8; ++nt)
      bias[m][nt] = bptr[m][nt * 16 + hn];

  // ---- residue validity flags (any atom valid) ----
  float resflag[4];
  #pragma unroll
  for (int j = 0; j < 4; ++j) {
    int any = mask[base + 4 * j] | mask[base + 4 * j + 1] |
              mask[base + 4 * j + 2] | mask[base + 4 * j + 3];
    resflag[j] = any ? 1.0f : 0.0f;
  }

  // ---- load X tile (fp32) into WMMA A-fragments (f16) ----
  // A operand, contiguous k order chosen here (same as B): lane m = hn; element i -> K = half*16 + i
  v16h a[4];
  {
    const float* xrow = x + (size_t)(base + hn) * CDIM;
    #pragma unroll
    for (int kc = 0; kc < 4; ++kc) {
      union { v16h v; _Float16 h[16]; } u;
      #pragma unroll
      for (int g = 0; g < 2; ++g) {
        const float* p = xrow + kc * 32 + half * 16 + g * 8;
        #pragma unroll
        for (int j = 0; j < 8; ++j) u.h[g * 8 + j] = (_Float16)p[j];
      }
      a[kc] = u.v;
    }
  }

  // ---- Q/K/V projections: 3 mats x 8 n-tiles x 4 k-steps of WMMA ----
  // B operand, contiguous k order chosen here (same as A): lane n = hn; element i -> K = half*16 + i
  #pragma unroll
  for (int m = 0; m < 3; ++m) {
    const _Float16* Wm = Wt + m * CDIM * CDIM;
    _Float16* dst = (m == 0) ? qb : (m == 1) ? kb : vb;
    #pragma unroll
    for (int nt = 0; nt < 8; ++nt) {
      v8f acc;
      #pragma unroll
      for (int r = 0; r < 8; ++r) acc[r] = bias[m][nt];
      #pragma unroll
      for (int kc = 0; kc < 4; ++kc) {
        const v16h b = *(const v16h*)(Wm + (size_t)(nt * 16 + hn) * CDIM +
                                      kc * 32 + half * 16);
        acc = wmma_f16(a[kc], b, acc);
      }
      // C/D layout: VGPR r -> token r + 8*half, column n = nt*16+hn
      #pragma unroll
      for (int r = 0; r < 8; ++r) {
        int tok = r + half * 8;
        dst[tok * CDIM + nt * 16 + hn] = (_Float16)acc[r];
      }
    }
  }

  asm volatile("s_wait_dscnt 0" ::: "memory");

  // ---- per-residue 4x4 softmax attention; lane = (token, head) ----
  const float scale = 0.17677669529663687f;  // 32^-0.5
  #pragma unroll
  for (int it = 0; it < 2; ++it) {
    const int t  = hn;
    const int h  = half + 2 * it;
    const int r0 = (t >> 2) << 2;  // first atom of this residue

    float qv[HD];
    {
      const _Float16* qrow = qb + t * CDIM + h * HD;
      union { v16h v; _Float16 hh[16]; } u0, u1;
      u0.v = *(const v16h*)(qrow);
      u1.v = *(const v16h*)(qrow + 16);
      #pragma unroll
      for (int d = 0; d < 16; ++d) {
        qv[d] = (float)u0.hh[d];
        qv[16 + d] = (float)u1.hh[d];
      }
    }

    float s[4];
    #pragma unroll
    for (int e = 0; e < 4; ++e) {
      const _Float16* krow = kb + (r0 + e) * CDIM + h * HD;
      union { v16h v; _Float16 hh[16]; } u0, u1;
      u0.v = *(const v16h*)(krow);
      u1.v = *(const v16h*)(krow + 16);
      float dot = 0.0f;
      #pragma unroll
      for (int d = 0; d < 16; ++d)
        dot += qv[d] * (float)u0.hh[d] + qv[16 + d] * (float)u1.hh[d];
      s[e] = (mask[base + r0 + e] != 0) ? dot * scale : -1.0e30f;
    }

    float mx = fmaxf(fmaxf(s[0], s[1]), fmaxf(s[2], s[3]));
    float p[4], sum = 0.0f;
    #pragma unroll
    for (int e = 0; e < 4; ++e) { p[e] = __expf(s[e] - mx); sum += p[e]; }
    float inv = (mx < -1.0e29f) ? 0.0f : (1.0f / sum);  // all-masked row -> 0

    float o[HD];
    #pragma unroll
    for (int d = 0; d < HD; ++d) o[d] = 0.0f;
    #pragma unroll
    for (int e = 0; e < 4; ++e) {
      const float pe = p[e] * inv;
      const _Float16* vrow = vb + (r0 + e) * CDIM + h * HD;
      union { v16h v; _Float16 hh[16]; } u0, u1;
      u0.v = *(const v16h*)(vrow);
      u1.v = *(const v16h*)(vrow + 16);
      #pragma unroll
      for (int d = 0; d < 16; ++d) {
        o[d] += pe * (float)u0.hh[d];
        o[16 + d] += pe * (float)u1.hh[d];
      }
    }
    _Float16* arow = ab + t * CDIM + h * HD;
    #pragma unroll
    for (int d = 0; d < HD; ++d) arow[d] = (_Float16)o[d];
  }

  asm volatile("s_wait_dscnt 0" ::: "memory");

  // ---- output projection: attended @ Wo + bo, residue-masked store ----
  v16h af[4];
  {
    const _Float16* arow = ab + hn * CDIM;
    #pragma unroll
    for (int kc = 0; kc < 4; ++kc) {
      union { v16h v; v8h h8[2]; } u;
      u.h8[0] = *(const v8h*)(arow + kc * 32 + half * 16);
      u.h8[1] = *(const v8h*)(arow + kc * 32 + half * 16 + 8);
      af[kc] = u.v;
    }
  }
  const _Float16* Wom = Wt + 3 * CDIM * CDIM;
  #pragma unroll
  for (int nt = 0; nt < 8; ++nt) {
    v8f acc;
    #pragma unroll
    for (int r = 0; r < 8; ++r) acc[r] = bias[3][nt];
    #pragma unroll
    for (int kc = 0; kc < 4; ++kc) {
      const v16h b = *(const v16h*)(Wom + (size_t)(nt * 16 + hn) * CDIM +
                                    kc * 32 + half * 16);
      acc = wmma_f16(af[kc], b, acc);
    }
    #pragma unroll
    for (int r = 0; r < 8; ++r) {
      int tok = r + half * 8;
      out[(size_t)(base + tok) * CDIM + nt * 16 + hn] =
          acc[r] * resflag[tok >> 2];
    }
  }
}

extern "C" void kernel_launch(void* const* d_in, const int* in_sizes, int n_in,
                              void* d_out, int out_size, void* d_ws,
                              size_t ws_size, hipStream_t stream) {
  const float* x  = (const float*)d_in[0];
  const int*   mk = (const int*)d_in[1];
  const float* Wq = (const float*)d_in[2];
  const float* bq = (const float*)d_in[3];
  const float* Wk = (const float*)d_in[4];
  const float* bk = (const float*)d_in[5];
  const float* Wv = (const float*)d_in[6];
  const float* bv = (const float*)d_in[7];
  const float* Wo = (const float*)d_in[8];
  const float* bo = (const float*)d_in[9];
  float*     out  = (float*)d_out;
  _Float16*  Wt   = (_Float16*)d_ws;  // 4 * 128*128 f16 = 128 KB scratch

  prep_weights_kernel<<<(4 * CDIM * CDIM + 255) / 256, 256, 0, stream>>>(
      Wq, Wk, Wv, Wo, Wt);

  const int tokens = in_sizes[0] / CDIM;       // B*L*A = 131072
  const int ntiles = tokens / TILE;            // 8192 tiles of 16 tokens
  const int blocks = (ntiles + WAVES - 1) / WAVES;
  fused_local_attn<<<blocks, WAVES * 32, 0, stream>>>(
      x, mk, Wt, bq, bk, bv, bo, out, ntiles);
}
